// hDCELoss_17068200035042
// MI455X (gfx1250) — matrix operand order repaired and checked by reading, not compiled, not run on hardware
//
#include <hip/hip_runtime.h>

// ---------------------------------------------------------------------------
// Problem constants (from reference setup_inputs)
// ---------------------------------------------------------------------------
#define BATCH   4
#define SEQL    2048
#define CDIM    256
#define KCODES  8192
#define MROWS   (BATCH * SEQL)      // 8192
#define NUM_HARD 16
#define INV_TEMP (1.0f / 0.07f)

typedef __attribute__((ext_vector_type(8)))  float  v8f;
typedef __attribute__((ext_vector_type(8)))  __bf16 bf16x8;
typedef __attribute__((ext_vector_type(16))) __bf16 bf16x16;

// ---------------------------------------------------------------------------
// Workspace layout (bytes)
// ---------------------------------------------------------------------------
#define OFF_CB16   ((size_t)0)                                   // K*C*2  = 4 MB
#define OFF_Q      (OFF_CB16 + (size_t)KCODES * CDIM * 2)        // M*C*4  = 8 MB
#define OFF_PE16   (OFF_Q    + (size_t)MROWS  * CDIM * 4)        // M*C*2  = 4 MB
#define OFF_CNORM2 (OFF_PE16 + (size_t)MROWS  * CDIM * 2)        // K*4
#define OFF_INVN   (OFF_CNORM2 + (size_t)KCODES * 4)             // K*4
#define OFF_HARD   (OFF_INVN   + (size_t)KCODES * 4)             // M*16*4
#define OFF_RLOSS  (OFF_HARD   + (size_t)MROWS * NUM_HARD * 4)   // M*4

// ---------------------------------------------------------------------------
// Helpers
// ---------------------------------------------------------------------------
__device__ __forceinline__ float blockReduceSum256(float v, float* red8) {
    #pragma unroll
    for (int m = 16; m >= 1; m >>= 1) v += __shfl_xor(v, m, 32);
    if ((threadIdx.x & 31) == 0) red8[threadIdx.x >> 5] = v;
    __syncthreads();
    float s = 0.f;
    #pragma unroll
    for (int i = 0; i < 8; ++i) s += red8[i];
    __syncthreads();
    return s;
}

// ---------------------------------------------------------------------------
// Kernel 1: codebook prep — bf16 copy, squared norms, inverse norms
// ---------------------------------------------------------------------------
__global__ void __launch_bounds__(256) prep_codebook(
    const float* __restrict__ cb, __bf16* __restrict__ cb16,
    float* __restrict__ cnorm2, float* __restrict__ invn)
{
    __shared__ float red[8];
    const int k = blockIdx.x, c = threadIdx.x;
    const float v = cb[(size_t)k * CDIM + c];
    cb16[(size_t)k * CDIM + c] = (__bf16)v;
    const float ss = blockReduceSum256(v * v, red);
    if (threadIdx.x == 0) {
        cnorm2[k] = ss;
        invn[k]   = 1.0f / fmaxf(sqrtf(ss), 1e-12f);
    }
}

// ---------------------------------------------------------------------------
// Kernel 2: row prep — q = l2norm(feat) (fp32), positive embedding in bf16
// ---------------------------------------------------------------------------
__global__ void __launch_bounds__(256) prep_rows(
    const float* __restrict__ feat, const float* __restrict__ cb,
    const int* __restrict__ pidx,
    float* __restrict__ q, __bf16* __restrict__ pe16)
{
    __shared__ float red[8];
    const int r = blockIdx.x, c = threadIdx.x;
    const float f = feat[(size_t)r * CDIM + c];
    const float ss = blockReduceSum256(f * f, red);
    const float qi = 1.0f / fmaxf(sqrtf(ss), 1e-12f);
    q[(size_t)r * CDIM + c] = f * qi;
    int p = pidx[r];
    p = p < 0 ? 0 : (p > KCODES - 1 ? KCODES - 1 : p);
    pe16[(size_t)r * CDIM + c] = (__bf16)cb[(size_t)p * CDIM + c];
}

// ---------------------------------------------------------------------------
// Kernel 3: hard-negative mining.
// Block = 16 rows x all 8192 codes, 8 waves. A fragments (loop-invariant)
// hoisted into 64 VGPRs; per 16x16 tile: one clause of 16 global b128 loads
// for B, then 8 back-to-back v_wmma_f32_16x16x32_bf16 on the accumulator
// chain. Rank key: cnorm2[n] - 2*dot (row-constant ||pe||^2 dropped).
// Register top-17 per lane, LDS merge, emit indices 1..16 (0 = positive).
// ---------------------------------------------------------------------------
__global__ void __launch_bounds__(256) mine_hard(
    const __bf16* __restrict__ pe16, const __bf16* __restrict__ cb16,
    const float* __restrict__ cnorm2, int* __restrict__ hard)
{
    __shared__ float dtile[8][16][20];   // per-wave transpose; stride 20:
                                         // 16B-aligned rows + conflict-free
    __shared__ float topd[8][32][17];
    __shared__ int   topi[8][32][17];

    const int tid  = threadIdx.x;
    const int wave = tid >> 5;
    const int lane = tid & 31;
    const int nloc = tid & 15;          // A row / B column within tile
    const int half = (tid >> 4) & 1;    // lane half per ISA fragment layout
    const int rowbase = blockIdx.x * 16;

    // ---- A fragments: invariant over the whole column sweep; keep in VGPRs.
    // 16-bit A 16x32 layout: lane half selects K-phase; two 16B chunks at
    // k = kk+8*half and kk+16+8*half of row 'nloc'.
    bf16x16 afr[8];
    #pragma unroll
    for (int stp = 0; stp < 8; ++stp) {
        const __bf16* ap = &pe16[(size_t)(rowbase + nloc) * CDIM
                                 + stp * 32 + 8 * half];
        const bf16x8 alo = *(const bf16x8*)ap;
        const bf16x8 ahi = *(const bf16x8*)(ap + 16);
        afr[stp] = __builtin_shufflevector(alo, ahi,
            0,1,2,3,4,5,6,7,8,9,10,11,12,13,14,15);
    }

    float bd[17]; int bi[17];
    #pragma unroll
    for (int s = 0; s < 17; ++s) { bd[s] = 3.0e38f; bi[s] = -1; }

    for (int it = 0; it < 64; ++it) {
        const int colbase = it * 128 + wave * 16;
        if (it < 63) // warm L2 for the next sweep step's B rows
            __builtin_prefetch(&cb16[(size_t)(colbase + 128 + nloc) * CDIM], 0, 1);

        // Candidate-key norms for this tile: 8 contiguous floats per lane.
        const float4* cnp = (const float4*)(cnorm2 + colbase + half * 8);
        const float4 cn0 = cnp[0], cn1 = cnp[1];

        // ---- Batch all B fragments first (one load clause), then WMMA chain.
        // B column n = codebook row; lane half selects K-phase: two 16B chunks
        // at k = kk+8*half and kk+16+8*half (same order as A).
        bf16x16 bfr[8];
        #pragma unroll
        for (int stp = 0; stp < 8; ++stp) {
            const __bf16* bp = &cb16[(size_t)(colbase + nloc) * CDIM
                                     + stp * 32 + 8 * half];
            const bf16x8 blo = *(const bf16x8*)bp;
            const bf16x8 bhi = *(const bf16x8*)(bp + 16);
            bfr[stp] = __builtin_shufflevector(blo, bhi,
                0,1,2,3,4,5,6,7,8,9,10,11,12,13,14,15);
        }

        v8f acc = {0.f, 0.f, 0.f, 0.f, 0.f, 0.f, 0.f, 0.f};
        #pragma unroll
        for (int stp = 0; stp < 8; ++stp)
            acc = __builtin_amdgcn_wmma_f32_16x16x32_bf16(
                false, afr[stp], false, bfr[stp], (short)0, acc, false, false);

        // Transpose 16x16 accumulator through LDS so each lane owns a row.
        #pragma unroll
        for (int r = 0; r < 8; ++r)
            dtile[wave][r + 8 * half][nloc] = acc[r];
        __builtin_amdgcn_wave_barrier();   // same-wave LDS ops are in-order

        // Lane owns row 'nloc'; its 8 candidate dots are contiguous -> b128s.
        const float4* dp = (const float4*)&dtile[wave][nloc][half * 8];
        const float4 d0 = dp[0], d1 = dp[1];
        const float dv[8]  = {d0.x, d0.y, d0.z, d0.w, d1.x, d1.y, d1.z, d1.w};
        const float cnv[8] = {cn0.x, cn0.y, cn0.z, cn0.w, cn1.x, cn1.y, cn1.z, cn1.w};

        #pragma unroll
        for (int j = 0; j < 8; ++j) {
            const int   n   = colbase + half * 8 + j;
            const float key = cnv[j] - 2.0f * dv[j];   // d2 minus row const
            if (key < bd[16]) {
                bd[16] = key; bi[16] = n;
                #pragma unroll
                for (int s = 16; s > 0; --s)
                    if (bd[s] < bd[s - 1]) {
                        float td = bd[s]; bd[s] = bd[s-1]; bd[s-1] = td;
                        int   ti = bi[s]; bi[s] = bi[s-1]; bi[s-1] = ti;
                    }
            }
        }
        __builtin_amdgcn_wave_barrier();
    }

    // Publish per-lane partial lists, then merge 16 lists per row.
    #pragma unroll
    for (int s = 0; s < 17; ++s) { topd[wave][lane][s] = bd[s]; topi[wave][lane][s] = bi[s]; }
    __syncthreads();

    if (tid < 16) {
        float md[17]; int mi[17];
        #pragma unroll
        for (int s = 0; s < 17; ++s) { md[s] = 3.0e38f; mi[s] = -1; }
        for (int w = 0; w < 8; ++w)
            for (int l2 = 0; l2 < 2; ++l2) {
                const int ln = tid + 16 * l2;
                for (int k = 0; k < 17; ++k) {
                    const float dval = topd[w][ln][k];
                    if (dval >= md[16]) break;     // source lists are sorted
                    const int iv = topi[w][ln][k];
                    md[16] = dval; mi[16] = iv;
                    #pragma unroll
                    for (int s = 16; s > 0; --s)
                        if (md[s] < md[s - 1]) {
                            float td = md[s]; md[s] = md[s-1]; md[s-1] = td;
                            int   ti = mi[s]; mi[s] = mi[s-1]; mi[s-1] = ti;
                        }
                }
            }
        // entry 0 is the positive itself (d2 ~ 0) — emit 1..16
        #pragma unroll
        for (int j = 0; j < NUM_HARD; ++j)
            hard[(size_t)(rowbase + tid) * NUM_HARD + j] = mi[j + 1];
    }
}

// ---------------------------------------------------------------------------
// Kernel 4: per-row logits (pos + 16 hard negs, fp32) + logsumexp loss.
// One wave per row; lane owns 8 contiguous channels -> b128 loads.
// ---------------------------------------------------------------------------
__global__ void __launch_bounds__(256) neg_logits_loss(
    const float* __restrict__ q, const float* __restrict__ cb,
    const float* __restrict__ invn, const int* __restrict__ pidx,
    const int* __restrict__ hard, float* __restrict__ rowloss)
{
    const int wave = threadIdx.x >> 5;
    const int lane = threadIdx.x & 31;
    const int row  = blockIdx.x * 8 + wave;

    int tgt[17];
    {
        int p = pidx[row];
        tgt[0] = p < 0 ? 0 : (p > KCODES - 1 ? KCODES - 1 : p);
    }
    #pragma unroll
    for (int j = 0; j < NUM_HARD; ++j) {
        int h = hard[(size_t)row * NUM_HARD + j];
        tgt[j + 1] = h < 0 ? 0 : (h > KCODES - 1 ? KCODES - 1 : h);
    }

    const float4* qv = (const float4*)(q + (size_t)row * CDIM + lane * 8);
    const float4 q0 = qv[0], q1 = qv[1];

    float logits[17];
    #pragma unroll
    for (int j = 0; j < 17; ++j) {
        const float4* cv = (const float4*)(cb + (size_t)tgt[j] * CDIM + lane * 8);
        const float4 c0 = cv[0], c1 = cv[1];
        float p = q0.x * c0.x + q0.y * c0.y + q0.z * c0.z + q0.w * c0.w
                + q1.x * c1.x + q1.y * c1.y + q1.z * c1.z + q1.w * c1.w;
        #pragma unroll
        for (int m = 16; m >= 1; m >>= 1) p += __shfl_xor(p, m, 32);
        logits[j] = p * invn[tgt[j]] * INV_TEMP;
    }

    float mx = logits[0];
    #pragma unroll
    for (int j = 1; j < 17; ++j) mx = fmaxf(mx, logits[j]);
    float se = 0.f;
    #pragma unroll
    for (int j = 0; j < 17; ++j) se += expf(logits[j] - mx);
    const float lse = mx + logf(se);
    if (lane == 0) rowloss[row] = lse - logits[0];
}

// ---------------------------------------------------------------------------
// Kernel 5: deterministic mean over 8192 row losses.
// ---------------------------------------------------------------------------
__global__ void __launch_bounds__(256) reduce_loss(
    const float* __restrict__ rowloss, float* __restrict__ out)
{
    __shared__ float red[256];
    float s = 0.f;
    for (int i = threadIdx.x; i < MROWS; i += 256) s += rowloss[i];
    red[threadIdx.x] = s;
    __syncthreads();
    for (int k = 128; k > 0; k >>= 1) {
        if (threadIdx.x < k) red[threadIdx.x] += red[threadIdx.x + k];
        __syncthreads();
    }
    if (threadIdx.x == 0) out[0] = red[0] * (1.0f / (float)MROWS);
}

// ---------------------------------------------------------------------------
extern "C" void kernel_launch(void* const* d_in, const int* in_sizes, int n_in,
                              void* d_out, int out_size, void* d_ws, size_t ws_size,
                              hipStream_t stream) {
    const float* feat = (const float*)d_in[0];   // (B,L,C) f32
    const float* cb   = (const float*)d_in[1];   // (K,C)   f32
    const int*   pidx = (const int*)  d_in[2];   // (B,L)   i32
    float* out = (float*)d_out;

    char* ws = (char*)d_ws;
    __bf16* cb16   = (__bf16*)(ws + OFF_CB16);
    float*  q      = (float*) (ws + OFF_Q);
    __bf16* pe16   = (__bf16*)(ws + OFF_PE16);
    float*  cnorm2 = (float*) (ws + OFF_CNORM2);
    float*  invn   = (float*) (ws + OFF_INVN);
    int*    hard   = (int*)   (ws + OFF_HARD);
    float*  rloss  = (float*) (ws + OFF_RLOSS);

    prep_codebook<<<KCODES, 256, 0, stream>>>(cb, cb16, cnorm2, invn);
    prep_rows<<<MROWS, 256, 0, stream>>>(feat, cb, pidx, q, pe16);
    mine_hard<<<MROWS / 16, 256, 0, stream>>>(pe16, cb16, cnorm2, hard);
    neg_logits_loss<<<MROWS / 8, 256, 0, stream>>>(q, cb, invn, pidx, hard, rloss);
    reduce_loss<<<1, 256, 0, stream>>>(rloss, out);
}
